// LFSSBlock_72370198938170
// MI455X (gfx1250) — compile-verified, matrix operand order checked by reading, not run on hardware
//
#include <hip/hip_runtime.h>
#include <hip/hip_bf16.h>
#include <math.h>

// ---------------------------------------------------------------------------
// Problem constants (fixed by reference: B0=4, H0=W0=64, C=96)
// ---------------------------------------------------------------------------
#define B0   4
#define H0   64
#define W0   64
#define CCH  96
#define DI   192          // 2*C
#define KDIR 4
#define NST  16
#define RRK  6            // dt_rank = ceil(96/16)
#define LL   4096         // H0*W0
#define ROWS (B0*LL)      // 16384

typedef __bf16 bf16;
typedef __attribute__((ext_vector_type(8)))  bf16  v8bf;
typedef __attribute__((ext_vector_type(16))) bf16  v16bf;
typedef __attribute__((ext_vector_type(8)))  float v8f;

#define AS1 __attribute__((address_space(1)))
#define AS3 __attribute__((address_space(3)))

#if __has_builtin(__builtin_amdgcn_global_load_async_to_lds_b32) && \
    __has_builtin(__builtin_amdgcn_s_wait_asynccnt)
#define HAVE_ASYNC_LDS 1
#else
#define HAVE_ASYNC_LDS 0
#endif

__device__ __forceinline__ float dev_silu(float x)  { return x / (1.f + __expf(-x)); }
__device__ __forceinline__ float dev_softplus(float x) { return x > 20.f ? x : log1pf(__expf(x)); }
__device__ __forceinline__ float dev_gelu(float x)  { return 0.5f * x * (1.f + erff(x * 0.70710678118654752f)); }

// ---------------------------------------------------------------------------
// Generic strided GEMM:  C[m,n] = sum_k A[m,k] * W[n,k] (+bias[n])
// bf16 WMMA 16x16x32, f32 accumulate.
// Block tile 128x64x32, 8 waves, 32x32 wave tile -> 4 WMMAs per K-step.
// ---------------------------------------------------------------------------
#define BM 128
#define BN 64
#define BK 32
#define BKP 40   // padded LDS row stride (bf16 elems): 80B, 16B aligned

__device__ __forceinline__ v16bf frag_from_lds(const bf16* p)
{
    v8bf lo = *(const v8bf*)(p);
    v8bf hi = *(const v8bf*)(p + 16);
    return __builtin_shufflevector(lo, hi, 0,1,2,3,4,5,6,7,8,9,10,11,12,13,14,15);
}

__global__ __launch_bounds__(256)
void k_gemm_bf16(const float* __restrict__ A, long long sAm, long long sAk,
                 const float* __restrict__ W, long long sWn, long long sWk,
                 const float* __restrict__ bias,
                 float* __restrict__ C, long long sCm, long long sCn,
                 int M, int N, int Kd)
{
    __shared__ __align__(16) bf16 a_s[BM * BKP];
    __shared__ __align__(16) bf16 b_s[BN * BKP];

    const int tid  = threadIdx.x;
    const int m0   = blockIdx.y * BM;
    const int n0   = blockIdx.x * BN;
    const int wave = tid >> 5;
    const int lane = tid & 31;
    const int wm   = (wave & 3) * 32;   // wave M sub-tile (32 rows)
    const int wn   = (wave >> 2) * 32;  // wave N sub-tile (32 cols)
    const int half = lane >> 4;
    const int row  = lane & 15;

    v8f acc00 = {}, acc01 = {}, acc10 = {}, acc11 = {};

    for (int kb = 0; kb < Kd; kb += BK) {
        // Prefetch next A K-tile into L2 (global_prefetch_b8)
        if (kb + BK < Kd && tid == 0)
            __builtin_prefetch(&A[(long long)m0 * sAm + (long long)(kb + BK) * sAk], 0, 0);

        // Cooperative fill (f32 -> bf16): A 128x32, B 64x32
        for (int i = tid; i < BM * BK; i += 256) {
            int r = i >> 5, k = i & 31;
            int gm = m0 + r, gk = kb + k;
            float av = (gm < M && gk < Kd) ? A[(long long)gm * sAm + (long long)gk * sAk] : 0.f;
            a_s[r * BKP + k] = (bf16)av;
        }
        for (int i = tid; i < BN * BK; i += 256) {
            int r = i >> 5, k = i & 31;
            int gn = n0 + r, gk = kb + k;
            float bv = (gn < N && gk < Kd) ? W[(long long)gn * sWn + (long long)gk * sWk] : 0.f;
            b_s[r * BKP + k] = (bf16)bv;
        }
        __syncthreads();

        // Fragments: lane<16 -> K 0..7 & 16..23 ; lane>=16 -> K 8..15 & 24..31
        v16bf fa0 = frag_from_lds(&a_s[(wm      + row) * BKP + half * 8]);
        v16bf fa1 = frag_from_lds(&a_s[(wm + 16 + row) * BKP + half * 8]);
        v16bf fb0 = frag_from_lds(&b_s[(wn      + row) * BKP + half * 8]);
        v16bf fb1 = frag_from_lds(&b_s[(wn + 16 + row) * BKP + half * 8]);

        acc00 = __builtin_amdgcn_wmma_f32_16x16x32_bf16(false, fa0, false, fb0, (short)0, acc00, false, false);
        acc01 = __builtin_amdgcn_wmma_f32_16x16x32_bf16(false, fa0, false, fb1, (short)0, acc01, false, false);
        acc10 = __builtin_amdgcn_wmma_f32_16x16x32_bf16(false, fa1, false, fb0, (short)0, acc10, false, false);
        acc11 = __builtin_amdgcn_wmma_f32_16x16x32_bf16(false, fa1, false, fb1, (short)0, acc11, false, false);
        __syncthreads();
    }

    // D layout: VGPR r, lanes 0-15 -> M=r, N=lane ; lanes 16-31 -> M=8+r, N=lane-16
    const int col = lane & 15;
    #pragma unroll
    for (int fm = 0; fm < 2; ++fm) {
        #pragma unroll
        for (int r = 0; r < 8; ++r) {
            int gm = m0 + wm + 16 * fm + r + 8 * half;
            if (gm >= M) continue;
            int gn0 = n0 + wn + col;
            int gn1 = gn0 + 16;
            float v0 = (fm ? acc10[r] : acc00[r]);
            float v1 = (fm ? acc11[r] : acc01[r]);
            if (gn0 < N) C[(long long)gm * sCm + (long long)gn0 * sCn] = v0 + (bias ? bias[gn0] : 0.f);
            if (gn1 < N) C[(long long)gm * sCm + (long long)gn1 * sCn] = v1 + (bias ? bias[gn1] : 0.f);
        }
    }
}

// ---------------------------------------------------------------------------
// LayerNorm over last dim (thread per row)
// ---------------------------------------------------------------------------
__global__ void k_ln(const float* __restrict__ x, const float* __restrict__ w,
                     const float* __restrict__ b, float* __restrict__ y,
                     int rows, int Cd)
{
    int r = blockIdx.x * blockDim.x + threadIdx.x;
    if (r >= rows) return;
    const float* xr = x + (long long)r * Cd;
    float m = 0.f;
    for (int c = 0; c < Cd; ++c) m += xr[c];
    m /= (float)Cd;
    float v = 0.f;
    for (int c = 0; c < Cd; ++c) { float t = xr[c] - m; v += t * t; }
    v /= (float)Cd;
    float inv = rsqrtf(v + 1e-5f);
    float* yr = y + (long long)r * Cd;
    for (int c = 0; c < Cd; ++c) yr[c] = (xr[c] - m) * inv * w[c] + b[c];
}

// ---------------------------------------------------------------------------
// Depthwise 3x3 SAME conv.  Input: NHWC with row stride inStride, channel
// offset inOff.  outLayout 0: (b, c, hw) channel-major; 1: (b, hw, c) NHWC.
// ---------------------------------------------------------------------------
__global__ void k_dwconv3(const float* __restrict__ x, int inStride, int inOff,
                          const float* __restrict__ wgt, const float* __restrict__ bias,
                          float* __restrict__ y, int Cch, int outLayout, int act)
{
    long long i = (long long)blockIdx.x * blockDim.x + threadIdx.x;
    long long total = (long long)B0 * Cch * LL;
    if (i >= total) return;
    int hw = (int)(i & (LL - 1));
    int h = hw >> 6, w = hw & 63;
    long long t = i >> 12;
    int c = (int)(t % Cch);
    int b = (int)(t / Cch);
    float s = bias[c];
    #pragma unroll
    for (int ky = 0; ky < 3; ++ky) {
        int hh = h + ky - 1;
        if (hh < 0 || hh >= H0) continue;
        #pragma unroll
        for (int kx = 0; kx < 3; ++kx) {
            int ww = w + kx - 1;
            if (ww < 0 || ww >= W0) continue;
            s += x[((long long)b * LL + hh * W0 + ww) * inStride + inOff + c] * wgt[c * 9 + ky * 3 + kx];
        }
    }
    if (act) s = dev_silu(s);
    long long oi = (outLayout == 0)
                 ? ((long long)(b * Cch + c)) * LL + hw
                 : ((long long)(b * LL + hw)) * Cch + c;
    y[oi] = s;
}

// ---------------------------------------------------------------------------
// Build 4 scan directions xs (B,4,DI,L) from xconv (B,DI,L)
// ---------------------------------------------------------------------------
__global__ void k_xs_build(const float* __restrict__ xconv, float* __restrict__ xs)
{
    long long i = (long long)blockIdx.x * blockDim.x + threadIdx.x;
    long long total = (long long)B0 * DI * LL;
    if (i >= total) return;
    int l = (int)(i & (LL - 1));
    long long t = i >> 12;
    int d = (int)(t % DI);
    int b = (int)(t / DI);
    int h = l >> 6, w = l & 63;
    int l2 = w * H0 + h;
    float v = xconv[i];
    long long bb = ((long long)b * KDIR) * DI + d;
    xs[(bb + 0 * DI) * LL + l]             = v;
    xs[(bb + 1 * DI) * LL + l2]            = v;
    xs[(bb + 2 * DI) * LL + (LL - 1 - l)]  = v;
    xs[(bb + 3 * DI) * LL + (LL - 1 - l2)] = v;
}

// ---------------------------------------------------------------------------
// delta = softplus(dts @ dt_projs_w + dt_projs_b)  ; dts = x_dbl rows 0..5
// ---------------------------------------------------------------------------
__global__ void k_delta(const float* __restrict__ xdbl, const float* __restrict__ dtw,
                        const float* __restrict__ dtb, float* __restrict__ delta)
{
    long long i = (long long)blockIdx.x * blockDim.x + threadIdx.x;
    long long total = (long long)B0 * KDIR * DI * LL;
    if (i >= total) return;
    int l = (int)(i & (LL - 1));
    long long t = i >> 12;
    int d  = (int)(t % DI);
    int bk = (int)(t / DI);
    int kd = (bk & 3) * DI + d;
    float s = dtb[kd];
    const float* wr = &dtw[(long long)kd * RRK];
    const float* xb = &xdbl[(long long)bk * 38 * LL];
    #pragma unroll
    for (int r = 0; r < RRK; ++r) s += xb[(long long)r * LL + l] * wr[r];
    delta[i] = dev_softplus(s);
}

// ---------------------------------------------------------------------------
// Selective scan: block per (b,k), thread per d. 16-state in registers.
// B/C vectors (x_dbl rows 6..37, i.e. 6+d for d in [0,32)) staged into LDS,
// via GLOBAL_LOAD_ASYNC_TO_LDS_B32 (ASYNCcnt) when available.
// ys written in-place over delta.
// ---------------------------------------------------------------------------
__global__ __launch_bounds__(DI)
void k_scan(const float* __restrict__ xs, float* __restrict__ dys,
            const float* __restrict__ xdbl, const float* __restrict__ A_logs,
            const float* __restrict__ Ds)
{
    const int bk = blockIdx.x;        // b*4+k
    const int d  = threadIdx.x;       // 0..191
    const int kd = (bk & 3) * DI + d;
    __shared__ float BCsh[32];        // [0:16)=B_t, [16:32)=C_t
    float h[NST], Ar[NST];
    #pragma unroll
    for (int n = 0; n < NST; ++n) {
        h[n] = 0.f;
        Ar[n] = -__expf(A_logs[(long long)kd * NST + n]);
    }
    const float Dv = Ds[kd];
    const long long baseU = ((long long)bk * DI + d) * LL;
    // lane d (<32) streams x_dbl row 6+d : rows 6..21 = B, 22..37 = C
    const float* gBC = xdbl + (long long)bk * 38 * LL + (long long)(6 + d) * LL;

    for (int l = 0; l < LL; ++l) {
        if (d < 32) {
#if HAVE_ASYNC_LDS
            __builtin_amdgcn_global_load_async_to_lds_b32(
                (AS1 int*)(gBC + l), (AS3 int*)&BCsh[d], 0, 0);
#else
            BCsh[d] = gBC[l];
#endif
        }
#if HAVE_ASYNC_LDS
        __builtin_amdgcn_s_wait_asynccnt(0);
#endif
        __syncthreads();
        float dl = dys[baseU + l];
        float u  = xs[baseU + l];
        float du = dl * u;
        float y  = 0.f;
        #pragma unroll
        for (int n = 0; n < NST; ++n) {
            h[n] = __expf(dl * Ar[n]) * h[n] + du * BCsh[n];
            y += h[n] * BCsh[16 + n];
        }
        dys[baseU + l] = y + u * Dv;
        __syncthreads();
    }
}

// ---------------------------------------------------------------------------
// Merge 4 directions -> ycomb (B,L,DI) row-major
// ---------------------------------------------------------------------------
__global__ void k_combine(const float* __restrict__ ys, float* __restrict__ ycomb)
{
    long long i = (long long)blockIdx.x * blockDim.x + threadIdx.x;
    long long total = (long long)B0 * DI * LL;
    if (i >= total) return;
    int l = (int)(i & (LL - 1));
    long long t = i >> 12;
    int d = (int)(t % DI);
    int b = (int)(t / DI);
    int h = l >> 6, w = l & 63;
    int l2 = w * H0 + h;
    long long bb = ((long long)b * KDIR) * DI + d;
    float v = ys[(bb + 0 * DI) * LL + l]
            + ys[(bb + 2 * DI) * LL + (LL - 1 - l)]
            + ys[(bb + 1 * DI) * LL + l2]
            + ys[(bb + 3 * DI) * LL + (LL - 1 - l2)];
    ycomb[((long long)b * LL + l) * DI + d] = v;
}

// ---------------------------------------------------------------------------
// out_norm LN over DI then * silu(z) ; z = xz[..., 192:384]
// ---------------------------------------------------------------------------
__global__ void k_ln_gate(const float* __restrict__ ycomb, const float* __restrict__ xz,
                          const float* __restrict__ w, const float* __restrict__ b,
                          float* __restrict__ yact, int rows)
{
    int r = blockIdx.x * blockDim.x + threadIdx.x;
    if (r >= rows) return;
    const float* xr = ycomb + (long long)r * DI;
    float m = 0.f;
    for (int c = 0; c < DI; ++c) m += xr[c];
    m /= (float)DI;
    float v = 0.f;
    for (int c = 0; c < DI; ++c) { float t = xr[c] - m; v += t * t; }
    v /= (float)DI;
    float inv = rsqrtf(v + 1e-5f);
    const float* zr = xz + (long long)r * (2 * DI) + DI;
    float* yr = yact + (long long)r * DI;
    for (int c = 0; c < DI; ++c)
        yr[c] = ((xr[c] - m) * inv * w[c] + b[c]) * dev_silu(zr[c]);
}

// gated = gelu(f2[:, :96]) * f2[:, 96:]
__global__ void k_gelu_gate(const float* __restrict__ f2, float* __restrict__ gated)
{
    long long i = (long long)blockIdx.x * blockDim.x + threadIdx.x;
    long long total = (long long)ROWS * CCH;
    if (i >= total) return;
    long long row = i / CCH;
    int c = (int)(i % CCH);
    float x1 = f2[row * DI + c];
    float x2 = f2[row * DI + CCH + c];
    gated[i] = dev_gelu(x1) * x2;
}

// o[i] = a[i]*sc[i%C] + b[i]
__global__ void k_scale_add(const float* __restrict__ a, const float* __restrict__ sc,
                            const float* __restrict__ b, float* __restrict__ o,
                            long long n, int Cc)
{
    long long i = (long long)blockIdx.x * blockDim.x + threadIdx.x;
    if (i >= n) return;
    int c = (int)(i % Cc);
    o[i] = a[i] * sc[c] + b[i];
}

// ---------------------------------------------------------------------------
// Orchestration
// ---------------------------------------------------------------------------
extern "C" void kernel_launch(void* const* d_in, const int* in_sizes, int n_in,
                              void* d_out, int out_size, void* d_ws, size_t ws_size,
                              hipStream_t stream)
{
    (void)in_sizes; (void)n_in; (void)out_size; (void)ws_size;
    const float* input     = (const float*)d_in[0];
    const float* ln1_w     = (const float*)d_in[3];
    const float* ln1_b     = (const float*)d_in[4];
    const float* skip1     = (const float*)d_in[5];
    const float* skip2     = (const float*)d_in[6];
    const float* ln2_w     = (const float*)d_in[7];
    const float* ln2_b     = (const float*)d_in[8];
    const float* in_proj_w = (const float*)d_in[9];
    const float* conv_w    = (const float*)d_in[10];
    const float* conv_b    = (const float*)d_in[11];
    const float* x_proj_w  = (const float*)d_in[12];
    const float* dt_projs_w= (const float*)d_in[13];
    const float* dt_projs_b= (const float*)d_in[14];
    const float* A_logs    = (const float*)d_in[15];
    const float* Ds        = (const float*)d_in[16];
    const float* out_nw    = (const float*)d_in[17];
    const float* out_nb    = (const float*)d_in[18];
    const float* out_projw = (const float*)d_in[19];
    const float* ffn1_w    = (const float*)d_in[20];
    const float* ffn1_b    = (const float*)d_in[21];
    const float* ffn2_w    = (const float*)d_in[22];
    const float* ffn2_b    = (const float*)d_in[23];
    const float* ffn3_w    = (const float*)d_in[24];
    const float* ffn3_b    = (const float*)d_in[25];
    float* out = (float*)d_out;
    float* ws  = (float*)d_ws;

    // Workspace layout (floats), lifetime-based aliasing.
    size_t o = 0;
    float* xln   = ws + o; o += (size_t)ROWS * CCH;          // LN1 out, later LN2 out
    float* xz    = ws + o; o += (size_t)ROWS * 2 * DI;       // in_proj out (xx|z); later f1|f2
    float* xconv = ws + o; o += (size_t)B0 * DI * LL;        // conv+silu; later gated
    float* xs    = ws + o; o += (size_t)B0 * KDIR * DI * LL; // 4 directions; later f3
    float* xdbl  = ws + o; o += (size_t)B0 * KDIR * 38 * LL; // dts|Bs|Cs
    float* dys   = ws + o; o += (size_t)B0 * KDIR * DI * LL; // delta, then ys in-place
    float* ycomb = ws + o; o += (size_t)ROWS * DI;           // merged; later yproj
    float* yact  = ws + o; o += (size_t)ROWS * DI;           // gated LN output
    float* xres  = ws + o; o += (size_t)ROWS * CCH;          // post-SS2D residual
    float* f1    = xz;                                       // alias (xz dead after ln_gate)
    float* f2    = xz + (size_t)ROWS * DI;
    float* gated = xconv;                                    // alias
    float* yproj = ycomb;                                    // alias
    float* f3    = xs;                                       // alias

    const int TB = 256;
    #define GRID1D(n) dim3((unsigned)(((n) + TB - 1) / TB))

    // 1) LN1
    k_ln<<<GRID1D(ROWS), TB, 0, stream>>>(input, ln1_w, ln1_b, xln, ROWS, CCH);

    // 2) in_proj: (16384 x 384) = xln(16384x96) @ in_proj_w(384x96)^T
    k_gemm_bf16<<<dim3((2*DI + BN - 1)/BN, (ROWS + BM - 1)/BM), 256, 0, stream>>>(
        xln, CCH, 1, in_proj_w, CCH, 1, nullptr, xz, 2*DI, 1, ROWS, 2*DI, CCH);

    // 3) depthwise conv 3x3 + SiLU on xx half of xz -> xconv (B,DI,L)
    k_dwconv3<<<GRID1D((long long)B0*DI*LL), TB, 0, stream>>>(
        xz, 2*DI, 0, conv_w, conv_b, xconv, DI, /*outLayout=*/0, /*act=*/1);

    // 4) build 4 scan directions
    k_xs_build<<<GRID1D((long long)B0*DI*LL), TB, 0, stream>>>(xconv, xs);

    // 5) x_dbl = einsum('bkdl,kcd->bkcl'): 16 GEMMs, M=L, N=38, K=DI
    for (int bk = 0; bk < B0 * KDIR; ++bk) {
        const float* Abk = xs + (long long)bk * DI * LL;             // (d,l): A[m=l,k=d]
        const float* Wbk = x_proj_w + (long long)(bk & 3) * 38 * DI; // (c,d)
        float* Cbk = xdbl + (long long)bk * 38 * LL;                 // (c,l)
        k_gemm_bf16<<<dim3(1, LL / BM), 256, 0, stream>>>(
            Abk, 1, LL, Wbk, DI, 1, nullptr, Cbk, 1, LL, LL, 38, DI);
    }

    // 6) delta = softplus(dt projection)
    k_delta<<<GRID1D((long long)B0*KDIR*DI*LL), TB, 0, stream>>>(
        xdbl, dt_projs_w, dt_projs_b, dys);

    // 7) selective scan (ys overwrites delta)
    k_scan<<<dim3(B0 * KDIR), DI, 0, stream>>>(xs, dys, xdbl, A_logs, Ds);

    // 8) merge directions -> ycomb (B,L,DI)
    k_combine<<<GRID1D((long long)B0*DI*LL), TB, 0, stream>>>(dys, ycomb);

    // 9) out_norm LN + silu(z) gate -> yact
    k_ln_gate<<<GRID1D(ROWS), TB, 0, stream>>>(ycomb, xz, out_nw, out_nb, yact, ROWS);

    // 10) out_proj: (16384 x 96) = yact(16384x192) @ out_proj_w(96x192)^T
    k_gemm_bf16<<<dim3((CCH + BN - 1)/BN, (ROWS + BM - 1)/BM), 256, 0, stream>>>(
        yact, DI, 1, out_projw, DI, 1, nullptr, yproj, CCH, 1, ROWS, CCH, DI);

    // 11) residual: xres = input*skip1 + yproj
    k_scale_add<<<GRID1D((long long)ROWS*CCH), TB, 0, stream>>>(
        input, skip1, yproj, xres, (long long)ROWS*CCH, CCH);

    // 12) LN2
    k_ln<<<GRID1D(ROWS), TB, 0, stream>>>(xres, ln2_w, ln2_b, xln, ROWS, CCH);

    // 13) ffn1 (1x1 conv = GEMM): (16384 x 192) + bias
    k_gemm_bf16<<<dim3((DI + BN - 1)/BN, (ROWS + BM - 1)/BM), 256, 0, stream>>>(
        xln, CCH, 1, ffn1_w, CCH, 1, ffn1_b, f1, DI, 1, ROWS, DI, CCH);

    // 14) ffn2 depthwise 3x3 (NHWC in/out)
    k_dwconv3<<<GRID1D((long long)B0*DI*LL), TB, 0, stream>>>(
        f1, DI, 0, ffn2_w, ffn2_b, f2, DI, /*outLayout=*/1, /*act=*/0);

    // 15) gelu gate
    k_gelu_gate<<<GRID1D((long long)ROWS*CCH), TB, 0, stream>>>(f2, gated);

    // 16) ffn3 (1x1 conv = GEMM): (16384 x 96) + bias
    k_gemm_bf16<<<dim3((CCH + BN - 1)/BN, (ROWS + BM - 1)/BM), 256, 0, stream>>>(
        gated, CCH, 1, ffn3_w, CCH, 1, ffn3_b, f3, CCH, 1, ROWS, CCH, CCH);

    // 17) final residual -> out
    k_scale_add<<<GRID1D((long long)ROWS*CCH), TB, 0, stream>>>(
        xres, skip2, f3, out, (long long)ROWS*CCH, CCH);
}
